// Net_6107443494972
// MI455X (gfx1250) — compile-verified, matrix operand order checked by reading, not run on hardware
//
#include <hip/hip_runtime.h>

// PNA GNN for MI455X (gfx1250, wave32, WMMA).
// Message GEMM (E=320k, K=225->256 padded-by-segment, N=375->384) runs on
// v_wmma_f32_16x16x32_f16 with register-resident A (global_load_b128 direct to
// fragment layout) and fragment-major pre-swizzled B (2x global_load_b128 per
// fragment). Results scatter via atomics into L2-resident (120MB < 192MB L2)
// sum/sumsq/min/max aggregators; min/max use order-preserving float->u32
// encoding so integer atomics implement float min/max.

#define N_NODES  20000
#define N_EDGES  320000
#define N_GRAPHS 200
#define TOWERS   5
#define F_IN     75
#define F_OUT    15
#define EDGE_DIM 50
#define NCOLS    375        // TOWERS * F_IN (message width)
#define NTILES   24         // ceil(375/16)
#define XS       80         // padded node-feature row stride (multiple of 8)
#define POST_IN  975
#define EPS_STD  1e-5f
#define EPS_BN   1e-5f

// Fragment-major B: [kc][nt][lane][16 halves]; per-(kc) stride:
#define B_KC_STRIDE (NTILES * 32 * 16)   // 12288 halves
#define B_TOTAL     (8 * B_KC_STRIDE)    // 98304 halves per layer

typedef __attribute__((ext_vector_type(16))) _Float16 v16h;
typedef __attribute__((ext_vector_type(8)))  float    v8f;

union AB {
  uint4 q[2];
  v16h  v;
};

__device__ __forceinline__ unsigned enc_ord(float f) {
  unsigned u = __float_as_uint(f);
  return (u & 0x80000000u) ? ~u : (u | 0x80000000u);
}
__device__ __forceinline__ float dec_ord(unsigned u) {
  return __uint_as_float((u & 0x80000000u) ? (u & 0x7fffffffu) : ~u);
}

// A-row layout (K=256): [x_dst 0..74 |pad| x_src 80..154 |pad| e 160..234 |pad|0]
// Chunk g (8 halves = 16B) lies entirely in one segment because 80 % 8 == 0.
__device__ __forceinline__ uint4 chunk_load(const _Float16* __restrict__ s0,
                                            const _Float16* __restrict__ s1,
                                            const _Float16* __restrict__ s2, int g) {
  if (g < 10) return *(const uint4*)(s0 + g * 8);
  if (g < 20) return *(const uint4*)(s1 + g * 8 - 80);
  if (g < 30) return *(const uint4*)(s2 + g * 8 - 160);
  uint4 z;
  z.x = z.y = z.z = z.w = 0u;
  return z;
}

__global__ void k_zero_f32(float* p, int count) {
  int i = blockIdx.x * blockDim.x + threadIdx.x;
  if (i < count) p[i] = 0.0f;
}

__global__ void k_deg(const int* __restrict__ dst, float* __restrict__ deg) {
  int i = blockIdx.x * blockDim.x + threadIdx.x;
  if (i < N_EDGES) atomicAdd(&deg[dst[i]], 1.0f);
}

__global__ __launch_bounds__(256) void k_avglog(const float* __restrict__ deg,
                                                float* __restrict__ alog) {
  __shared__ float s[256];
  int n = blockIdx.x * 256 + threadIdx.x;
  s[threadIdx.x] = (n < N_NODES) ? logf(deg[n] + 1.0f) : 0.0f;
  __syncthreads();
  for (int o = 128; o > 0; o >>= 1) {
    if ((int)threadIdx.x < o) s[threadIdx.x] += s[threadIdx.x + o];
    __syncthreads();
  }
  if (threadIdx.x == 0) atomicAdd(alog, s[0]);
}

__global__ void k_scalers(const float* __restrict__ deg, const float* __restrict__ alog,
                          float* __restrict__ amp, float* __restrict__ att) {
  int n = blockIdx.x * blockDim.x + threadIdx.x;
  if (n >= N_NODES) return;
  float al = alog[0] * (1.0f / (float)N_NODES);
  float ld = logf(fmaxf(deg[n], 1.0f) + 1.0f);
  amp[n] = ld / al;
  att[n] = al / ld;
}

__global__ void k_gather_x(const int* __restrict__ x_idx, const float* __restrict__ node_emb,
                           float* __restrict__ xf, _Float16* __restrict__ xh) {
  int i = blockIdx.x * blockDim.x + threadIdx.x;  // over N_NODES*XS
  if (i >= N_NODES * XS) return;
  int n = i / XS, c = i % XS;
  float v = (c < F_IN) ? node_emb[(size_t)x_idx[n] * F_IN + c] : 0.0f;
  xf[i] = v;
  xh[i] = (_Float16)v;
}

// e_table[l][a][:] = edge_emb[a] @ W_edge[l] + b_edge[l]  (only 100 distinct attrs)
__global__ void k_etab(const float* __restrict__ edge_emb, const float* __restrict__ W_edge,
                       const float* __restrict__ b_edge, _Float16* __restrict__ etab) {
  int i = blockIdx.x * blockDim.x + threadIdx.x;  // over 2*100*XS
  if (i >= 2 * 100 * XS) return;
  int l = i / (100 * XS);
  int r = i % (100 * XS);
  int a = r / XS, c = r % XS;
  float v = 0.0f;
  if (c < F_IN) {
    v = b_edge[l * F_IN + c];
    for (int k = 0; k < EDGE_DIM; ++k)
      v += edge_emb[a * EDGE_DIM + k] * W_edge[((size_t)l * EDGE_DIM + k) * F_IN + c];
  }
  etab[i] = (_Float16)v;
}

// Pack W_pre into fragment-major f16 B: Wbf[l][kc][nt][lane][j].
// Element j of a lane's fragment holds B[K][col] with
//   K = kc*32 + j + (j>=8 ? 8 : 0) + (lane>=16 ? 8 : 0), col = nt*16 + lane%16,
// and the segment-padded K mapping: seg=K/80, ko=K%80 valid if ko<75,
// logical c = seg*75+ko (matches the A-row layout).
__global__ void k_wpre_pack(const float* __restrict__ W_pre, _Float16* __restrict__ Wbf) {
  int i = blockIdx.x * blockDim.x + threadIdx.x;  // over 2*B_TOTAL
  if (i >= 2 * B_TOTAL) return;
  int l = i / B_TOTAL;
  int r = i % B_TOTAL;
  int j = r & 15;
  int lane = (r >> 4) & 31;
  int nt = (r >> 9) % NTILES;
  int kc = (r >> 9) / NTILES;
  int hi = lane >> 4;
  int kk = j + ((j >= 8) ? 8 : 0) + (hi << 3);
  int K = kc * 32 + kk;
  int col = nt * 16 + (lane & 15);
  int seg = K / 80, ko = K % 80;
  float v = 0.0f;
  if (seg < 3 && ko < F_IN && col < NCOLS) {
    int c = seg * F_IN + ko;
    int t = col / F_IN, f = col % F_IN;
    v = W_pre[(((size_t)l * TOWERS + t) * (3 * F_IN) + c) * F_IN + f];
  }
  Wbf[i] = (_Float16)v;
}

__global__ void k_init_aggr(float* __restrict__ sum, float* __restrict__ sumsq,
                            unsigned* __restrict__ mnb, unsigned* __restrict__ mxb) {
  int i = blockIdx.x * blockDim.x + threadIdx.x;
  if (i >= N_NODES * NCOLS) return;
  sum[i] = 0.0f;
  sumsq[i] = 0.0f;
  mnb[i] = 0xFFFFFFFFu;  // encoded +inf sentinel
  mxb[i] = 0u;           // encoded -inf sentinel
}

// One wave per 16-edge tile. A fragments (8 x v16h = 64 VGPRs) loaded once via
// global_load_b128 directly in WMMA layout; loop over all 24 n-tiles doing
// 8 WMMAs each with fragment-major B loads; scatter D (row M = v+8*hi,
// col N = lane%16) with 4 atomics per element.
__global__ __launch_bounds__(256) void k_msg_wmma(
    const _Float16* __restrict__ xh, const _Float16* __restrict__ etab,
    const _Float16* __restrict__ Wbf,
    const int* __restrict__ src, const int* __restrict__ dst,
    const int* __restrict__ eattr, const float* __restrict__ bpre,
    float* __restrict__ sum, float* __restrict__ sumsq,
    unsigned* __restrict__ mnb, unsigned* __restrict__ mxb) {
  const int lane = threadIdx.x & 31;
  const int wave = threadIdx.x >> 5;
  const int m = lane & 15;
  const int hi = lane >> 4;
  const int tile = blockIdx.x * 8 + wave;
  const int e0 = tile * 16;

  const int dmine = dst[e0 + m];
  const int smine = src[e0 + m];
  const int amine = eattr[e0 + m];
  const _Float16* s0 = xh + (size_t)dmine * XS;
  const _Float16* s1 = xh + (size_t)smine * XS;
  const _Float16* s2 = etab + (size_t)amine * XS;

  // dst index of each of this lane's 8 output rows (full EXEC for shfl)
  int drow[8];
#pragma unroll
  for (int v = 0; v < 8; ++v) drow[v] = __shfl(dmine, (hi << 3) + v, 32);

  // Load all A fragments; chunk segment choice is wave-uniform (4kc+hi and
  // 4kc+2+hi never straddle segment boundaries 10/20/30).
  AB afr[8];
#pragma unroll
  for (int kc = 0; kc < 8; ++kc) {
    afr[kc].q[0] = chunk_load(s0, s1, s2, kc * 4 + hi);
    afr[kc].q[1] = chunk_load(s0, s1, s2, kc * 4 + 2 + hi);
  }

#pragma unroll 1
  for (int nt = 0; nt < NTILES; ++nt) {
    const _Float16* bbase = Wbf + ((size_t)nt * 32 + lane) * 16;
    if (nt + 1 < NTILES) __builtin_prefetch(bbase + 512, 0, 0);  // next n-tile
    v8f acc;
#pragma unroll
    for (int j = 0; j < 8; ++j) acc[j] = 0.0f;
#pragma unroll
    for (int kc = 0; kc < 8; ++kc) {
      AB b;
      const uint4* bq = (const uint4*)(bbase + (size_t)kc * B_KC_STRIDE);
      b.q[0] = bq[0];
      b.q[1] = bq[1];
      acc = __builtin_amdgcn_wmma_f32_16x16x32_f16(false, afr[kc].v, false, b.v,
                                                   (short)0, acc, false, false);
    }
    int feat = nt * 16 + m;
    if (feat < NCOLS) {
      float bp = bpre[feat];  // b_pre flat [5][75] == feat index
#pragma unroll
      for (int v = 0; v < 8; ++v) {
        float val = acc[v] + bp;
        size_t o = (size_t)drow[v] * NCOLS + feat;
        atomicAdd(&sum[o], val);
        atomicAdd(&sumsq[o], val * val);
        unsigned u = enc_ord(val);
        atomicMax(&mxb[o], u);
        atomicMin(&mnb[o], u);
      }
    }
  }
}

// One wave per (node,tower): build h2[975] in LDS, then 15 shfl-reduced dots.
__global__ __launch_bounds__(256) void k_post(
    const float* __restrict__ xf,
    const float* __restrict__ sum, const float* __restrict__ sumsq,
    const unsigned* __restrict__ mnb, const unsigned* __restrict__ mxb,
    const float* __restrict__ deg, const float* __restrict__ amp,
    const float* __restrict__ att,
    const float* __restrict__ Wpost, const float* __restrict__ bpost,
    float* __restrict__ y1) {
  __shared__ float h2s[8][992];
  const int lane = threadIdx.x & 31;
  const int wave = threadIdx.x >> 5;
  const int gw = blockIdx.x * 8 + wave;  // node*TOWERS + tower (exact grid)
  const int n = gw / TOWERS;
  const int t = gw % TOWERS;
  float* h2 = h2s[wave];
  const float dg = deg[n];
  const float invd = 1.0f / fmaxf(dg, 1.0f);
  const bool hasin = dg > 0.0f;
  const float am = amp[n], at = att[n];
  for (int c = lane; c < F_IN; c += 32) h2[c] = xf[(size_t)n * XS + c];
  for (int idx = lane; idx < 4 * F_IN; idx += 32) {
    int agg = idx / F_IN, f = idx % F_IN;
    size_t o = (size_t)n * NCOLS + t * F_IN + f;
    float a;
    if (agg == 0)      a = sum[o] * invd;                   // mean
    else if (agg == 1) a = hasin ? dec_ord(mnb[o]) : 0.0f;  // min
    else if (agg == 2) a = hasin ? dec_ord(mxb[o]) : 0.0f;  // max
    else {                                                  // std
      float mean = sum[o] * invd;
      float msq = sumsq[o] * invd;
      a = sqrtf(fmaxf(msq - mean * mean, 0.0f) + EPS_STD);
    }
    h2[F_IN + idx] = a;
    h2[F_IN + 4 * F_IN + idx] = a * am;
    h2[F_IN + 8 * F_IN + idx] = a * at;
  }
  __syncthreads();
  const float* Wt = Wpost + (size_t)t * POST_IN * F_OUT;
  for (int j = 0; j < F_OUT; ++j) {
    float p = 0.0f;
    for (int c = lane; c < POST_IN; c += 32) p += h2[c] * Wt[(size_t)c * F_OUT + j];
    for (int o = 16; o > 0; o >>= 1) p += __shfl_down(p, o, 32);
    if (lane == 0) y1[(size_t)n * F_IN + t * F_OUT + j] = p + bpost[t * F_OUT + j];
  }
}

__global__ void k_lin(const float* __restrict__ y1, const float* __restrict__ Wlin,
                      const float* __restrict__ blin, float* __restrict__ y2) {
  int i = blockIdx.x * blockDim.x + threadIdx.x;
  if (i >= N_NODES * F_IN) return;
  int n = i / F_IN, o = i % F_IN;
  float acc = blin[o];
  const float* row = y1 + (size_t)n * F_IN;
  for (int c = 0; c < F_IN; ++c) acc += row[c] * Wlin[c * F_IN + o];
  y2[i] = acc;
}

__global__ __launch_bounds__(256) void k_bnstat(const float* __restrict__ y2,
                                                float* __restrict__ bns,
                                                float* __restrict__ bnq) {
  __shared__ float ss[256], qq[256];
  int f = blockIdx.x >> 5;      // 75 features
  int chunk = blockIdx.x & 31;  // 32 chunks each
  float s = 0.0f, q = 0.0f;
  for (int n = chunk * 256 + threadIdx.x; n < N_NODES; n += 32 * 256) {
    float v = y2[(size_t)n * F_IN + f];
    s += v;
    q += v * v;
  }
  ss[threadIdx.x] = s;
  qq[threadIdx.x] = q;
  __syncthreads();
  for (int o = 128; o > 0; o >>= 1) {
    if ((int)threadIdx.x < o) {
      ss[threadIdx.x] += ss[threadIdx.x + o];
      qq[threadIdx.x] += qq[threadIdx.x + o];
    }
    __syncthreads();
  }
  if (threadIdx.x == 0) {
    atomicAdd(&bns[f], ss[0]);
    atomicAdd(&bnq[f], qq[0]);
  }
}

__global__ void k_bn_apply(const float* __restrict__ y2, const float* __restrict__ bns,
                           const float* __restrict__ bnq, const float* __restrict__ gamma,
                           const float* __restrict__ beta, float* __restrict__ xf,
                           _Float16* __restrict__ xh, int do_pool,
                           const int* __restrict__ batch, float* __restrict__ pooled) {
  int i = blockIdx.x * blockDim.x + threadIdx.x;  // over N_NODES*XS
  if (i >= N_NODES * XS) return;
  int n = i / XS, c = i % XS;
  if (c >= F_IN) {
    xf[i] = 0.0f;
    xh[i] = (_Float16)0.0f;
    return;
  }
  float mu = bns[c] * (1.0f / (float)N_NODES);
  float var = bnq[c] * (1.0f / (float)N_NODES) - mu * mu;  // biased variance
  float v = (y2[(size_t)n * F_IN + c] - mu) * rsqrtf(var + EPS_BN) * gamma[c] + beta[c];
  v = fmaxf(v, 0.0f);
  xf[i] = v;
  xh[i] = (_Float16)v;
  if (do_pool) atomicAdd(&pooled[(size_t)batch[n] * XS + c], v);
}

__global__ __launch_bounds__(256) void k_mlp(const float* __restrict__ pooled,
    const float* __restrict__ W1, const float* __restrict__ b1,
    const float* __restrict__ W2, const float* __restrict__ b2,
    const float* __restrict__ W3, const float* __restrict__ b3,
    float* __restrict__ g1, float* __restrict__ g2, float* __restrict__ out) {
  for (int i = threadIdx.x; i < N_GRAPHS * 50; i += 256) {
    int g = i / 50, o = i % 50;
    float a = b1[o];
    for (int c = 0; c < 75; ++c) a += pooled[(size_t)g * XS + c] * W1[c * 50 + o];
    g1[i] = fmaxf(a, 0.0f);
  }
  __syncthreads();
  for (int i = threadIdx.x; i < N_GRAPHS * 25; i += 256) {
    int g = i / 25, o = i % 25;
    float a = b2[o];
    for (int c = 0; c < 50; ++c) a += g1[g * 50 + c] * W2[c * 25 + o];
    g2[i] = fmaxf(a, 0.0f);
  }
  __syncthreads();
  for (int g = threadIdx.x; g < N_GRAPHS; g += 256) {
    float a = b3[0];
    for (int c = 0; c < 25; ++c) a += g2[g * 25 + c] * W3[c];
    out[g] = a;
  }
}

extern "C" void kernel_launch(void* const* d_in, const int* in_sizes, int n_in,
                              void* d_out, int out_size, void* d_ws, size_t ws_size,
                              hipStream_t stream) {
  (void)in_sizes; (void)n_in; (void)out_size; (void)ws_size;
  const int* x_idx = (const int*)d_in[0];
  const int* eidx = (const int*)d_in[1];
  const int* src = eidx;            // edge_index[0]
  const int* dst = eidx + N_EDGES;  // edge_index[1] (aggregate at dst)
  const int* eattr = (const int*)d_in[2];
  const int* batch = (const int*)d_in[3];
  const float* node_emb = (const float*)d_in[4];
  const float* edge_emb = (const float*)d_in[5];
  const float* W_edge = (const float*)d_in[6];
  const float* b_edge = (const float*)d_in[7];
  const float* W_pre = (const float*)d_in[8];
  const float* b_pre = (const float*)d_in[9];
  const float* W_post = (const float*)d_in[10];
  const float* b_post = (const float*)d_in[11];
  const float* W_lin = (const float*)d_in[12];
  const float* b_lin = (const float*)d_in[13];
  const float* bn_g = (const float*)d_in[14];
  const float* bn_b = (const float*)d_in[15];
  const float* W1 = (const float*)d_in[16];
  const float* b1 = (const float*)d_in[17];
  const float* W2 = (const float*)d_in[18];
  const float* b2 = (const float*)d_in[19];
  const float* W3 = (const float*)d_in[20];
  const float* b3 = (const float*)d_in[21];

  // Workspace carving (~143 MB total; aggregators fit in 192MB L2).
  char* base = (char*)d_ws;
  size_t off = 0;
  auto carve = [&](size_t bytes) -> char* {
    char* p = base + off;
    off += (bytes + 255) & ~(size_t)255;
    return p;
  };
  float*    xf     = (float*)carve((size_t)N_NODES * XS * 4);
  _Float16* xh     = (_Float16*)carve((size_t)N_NODES * XS * 2);
  _Float16* etab   = (_Float16*)carve((size_t)2 * 100 * XS * 2);
  _Float16* Wbf    = (_Float16*)carve((size_t)2 * B_TOTAL * 2);
  float*    sum    = (float*)carve((size_t)N_NODES * NCOLS * 4);
  float*    sumsq  = (float*)carve((size_t)N_NODES * NCOLS * 4);
  unsigned* mnb    = (unsigned*)carve((size_t)N_NODES * NCOLS * 4);
  unsigned* mxb    = (unsigned*)carve((size_t)N_NODES * NCOLS * 4);
  float*    deg    = (float*)carve((size_t)N_NODES * 4);
  float*    alog   = (float*)carve(256);
  float*    amp    = (float*)carve((size_t)N_NODES * 4);
  float*    att    = (float*)carve((size_t)N_NODES * 4);
  float*    y1     = (float*)carve((size_t)N_NODES * F_IN * 4);
  float*    y2     = (float*)carve((size_t)N_NODES * F_IN * 4);
  float*    bns    = (float*)carve(256);
  float*    bnq    = (float*)carve(256);
  float*    pooled = (float*)carve((size_t)N_GRAPHS * XS * 4);
  float*    g1     = (float*)carve((size_t)N_GRAPHS * 50 * 4);
  float*    g2     = (float*)carve((size_t)N_GRAPHS * 25 * 4);

  const int T = 256;
  k_zero_f32<<<(N_NODES + T - 1) / T, T, 0, stream>>>(deg, N_NODES);
  k_zero_f32<<<1, 32, 0, stream>>>(alog, 1);
  k_zero_f32<<<(N_GRAPHS * XS + T - 1) / T, T, 0, stream>>>(pooled, N_GRAPHS * XS);
  k_deg<<<(N_EDGES + T - 1) / T, T, 0, stream>>>(dst, deg);
  k_avglog<<<(N_NODES + 255) / 256, 256, 0, stream>>>(deg, alog);
  k_scalers<<<(N_NODES + T - 1) / T, T, 0, stream>>>(deg, alog, amp, att);
  k_gather_x<<<(N_NODES * XS) / T, T, 0, stream>>>(x_idx, node_emb, xf, xh);
  k_etab<<<(2 * 100 * XS + T - 1) / T, T, 0, stream>>>(edge_emb, W_edge, b_edge, etab);
  k_wpre_pack<<<(2 * B_TOTAL) / T, T, 0, stream>>>(W_pre, Wbf);

  for (int l = 0; l < 2; ++l) {
    k_init_aggr<<<(N_NODES * NCOLS + T - 1) / T, T, 0, stream>>>(sum, sumsq, mnb, mxb);
    k_msg_wmma<<<N_EDGES / 128, 256, 0, stream>>>(
        xh, etab + (size_t)l * 100 * XS, Wbf + (size_t)l * B_TOTAL,
        src, dst, eattr, b_pre + (size_t)l * TOWERS * F_IN,
        sum, sumsq, mnb, mxb);
    k_post<<<(N_NODES * TOWERS) / 8, 256, 0, stream>>>(
        xf, sum, sumsq, mnb, mxb, deg, amp, att,
        W_post + (size_t)l * TOWERS * POST_IN * F_OUT,
        b_post + (size_t)l * TOWERS * F_OUT, y1);
    k_lin<<<(N_NODES * F_IN + T - 1) / T, T, 0, stream>>>(
        y1, W_lin + (size_t)l * F_IN * F_IN, b_lin + (size_t)l * F_IN, y2);
    k_zero_f32<<<1, 128, 0, stream>>>(bns, F_IN);
    k_zero_f32<<<1, 128, 0, stream>>>(bnq, F_IN);
    k_bnstat<<<F_IN * 32, 256, 0, stream>>>(y2, bns, bnq);
    k_bn_apply<<<(N_NODES * XS) / T, T, 0, stream>>>(
        y2, bns, bnq, bn_g + l * F_IN, bn_b + l * F_IN, xf, xh,
        (l == 1) ? 1 : 0, batch, pooled);
  }
  k_mlp<<<1, 256, 0, stream>>>(pooled, W1, b1, W2, b2, W3, b3, g1, g2, (float*)d_out);
}
